// GFAPM_38242388803819
// MI455X (gfx1250) — matrix operand order repaired and checked by reading, not compiled, not run on hardware
//
#include <hip/hip_runtime.h>
#include <hip/hip_bf16.h>

// ---------------------------------------------------------------------------
// Problem constants (reference: B,C,H,W = 4,256,64,64)
// ---------------------------------------------------------------------------
#define BB   4
#define CC   256
#define C4   64
#define HW   4096
#define EPS  1e-5f

typedef __attribute__((ext_vector_type(16))) __bf16 v16bf;
typedef __attribute__((ext_vector_type(8)))  float  v8f;

union FragB { uint4 u[2]; v16bf f; };

__device__ __forceinline__ v8f wmma_bf16(v16bf a, v16bf b, v8f c) {
    // (neg_a, A, neg_b, B, c_mod, C, reuse_a, reuse_b)
    return __builtin_amdgcn_wmma_f32_16x16x32_bf16(false, a, false, b,
                                                   (short)0, c, false, false);
}

__device__ __forceinline__ unsigned short f2bf(float f) {
    unsigned u = __float_as_uint(f);
    u += 0x7FFFu + ((u >> 16) & 1u);          // round-to-nearest-even
    return (unsigned short)(u >> 16);
}

#if __has_builtin(__builtin_amdgcn_cvt_pk_bf16_f32)
typedef __attribute__((ext_vector_type(2))) __bf16 v2bf;
__device__ __forceinline__ unsigned f2bf2(float lo, float hi) {
    union { v2bf v; unsigned u; } cv;
    cv.v = __builtin_amdgcn_cvt_pk_bf16_f32(lo, hi);   // v_cvt_pk_bf16_f32
    return cv.u;
}
#else
__device__ __forceinline__ unsigned f2bf2(float lo, float hi) {
    return (unsigned)f2bf(lo) | ((unsigned)f2bf(hi) << 16);
}
#endif

// ---------------------------------------------------------------------------
// Stage 1a: 1x1 conv + BN + ReLU, OUTPUT TRANSPOSED  out[b][i][c]  (bf16)
// Used for Q and K. block = 256 threads = 64 channels x 4 spatial.
// ---------------------------------------------------------------------------
__global__ __launch_bounds__(256) void cbr_qk(
    const float* __restrict__ x,  const float* __restrict__ w,
    const float* __restrict__ bi, const float* __restrict__ g,
    const float* __restrict__ be, const float* __restrict__ mn,
    const float* __restrict__ vr, unsigned short* __restrict__ outT)
{
    const int c    = threadIdx.x & 63;
    const int isub = threadIdx.x >> 6;
    const int i    = blockIdx.x * 4 + isub;
    const int b    = blockIdx.y;

    const float* xp = x + (size_t)b * CC * HW + i;
    const float* wp = w + c * CC;
    float acc = 0.f;
#pragma unroll 8
    for (int ch = 0; ch < CC; ++ch)
        acc = fmaf(wp[ch], xp[(size_t)ch * HW], acc);

    const float sc = g[c] * rsqrtf(vr[c] + EPS);
    float y = fmaf(acc + bi[c], sc, be[c] - mn[c] * sc);
    y = fmaxf(y, 0.f);
    outT[((size_t)b * HW + i) * C4 + c] = f2bf(y);
}

// ---------------------------------------------------------------------------
// Stage 1b: 1x1 conv + BN + ReLU, NATURAL layout out[b][c][i] (bf16).  For V.
// ---------------------------------------------------------------------------
__global__ __launch_bounds__(256) void cbr_v(
    const float* __restrict__ x,  const float* __restrict__ w,
    const float* __restrict__ bi, const float* __restrict__ g,
    const float* __restrict__ be, const float* __restrict__ mn,
    const float* __restrict__ vr, unsigned short* __restrict__ outp)
{
    const int i = blockIdx.x * 256 + threadIdx.x;
    const int c = blockIdx.y;
    const int b = blockIdx.z;

    const float* xp = x + (size_t)b * CC * HW + i;
    const float* wp = w + c * CC;
    float acc = 0.f;
#pragma unroll 8
    for (int ch = 0; ch < CC; ++ch)
        acc = fmaf(wp[ch], xp[(size_t)ch * HW], acc);

    const float sc = g[c] * rsqrtf(vr[c] + EPS);
    float y = fmaf(acc + bi[c], sc, be[c] - mn[c] * sc);
    y = fmaxf(y, 0.f);
    outp[((size_t)b * CC + c) * HW + i] = f2bf(y);
}

// ---------------------------------------------------------------------------
// Stage 2: fused flash attention + residual.
//   WG = 4 waves, one (batch, 16-query tile). Wave w owns channels [64w,64w+64).
//   Per 64-j block (16 WMMAs / iter):
//     S^T (4 j-tiles x 16i) = Kt x Qt            -> 8 x wmma_bf16 (contract c=64)
//     fp32 online softmax over j                 -> VALU + 2 shuffles
//     P -> bf16 -> wave-private LDS (no barriers; LDS is in-order per wave)
//     O^T (4 c-tiles) += V x P^T (contract j=64) -> 8 x wmma_bf16
// ---------------------------------------------------------------------------
__global__ __launch_bounds__(128) void attn_fused(
    const unsigned short* __restrict__ qT,   // (B, HW, 64) bf16
    const unsigned short* __restrict__ kT,   // (B, HW, 64) bf16
    const unsigned short* __restrict__ vmat, // (B, 256, HW) bf16
    const float* __restrict__ x1r,           // residual (B,C,HW) f32
    const float* __restrict__ gamp,          // scalar gamma
    float* __restrict__ outp)                // (B,C,HW) f32
{
    // [wave][i][j], row stride 72 elems (144 B): 16B-aligned b128s, bank-spread
    __shared__ __align__(16) unsigned short pT[4][16 * 72];

    const int lane = threadIdx.x & 31;
    const int wv   = threadIdx.x >> 5;
    const int hf   = lane >> 4;     // which 16-lane half
    const int ln   = lane & 15;
    const int i0   = blockIdx.x * 16;
    const int b    = blockIdx.y;
    const float L2E = 1.44269504088896340736f;

    // --- Q B-fragments (K=c contraction), resident for the whole kernel ----
    // B layout (16-bit, 32x16): lane n = i, u0 -> K = 8*hf.., u1 -> K = 16+8*hf..
    const unsigned short* qrow = qT + ((size_t)b * HW + i0 + ln) * C4;
    FragB qb0, qb1;
    qb0.u[0] = *(const uint4*)(qrow + 8 * hf);
    qb0.u[1] = *(const uint4*)(qrow + 16 + 8 * hf);
    qb1.u[0] = *(const uint4*)(qrow + 32 + 8 * hf);
    qb1.u[1] = *(const uint4*)(qrow + 48 + 8 * hf);

    // K row base for this lane (A layout: lane m = j-within-tile)
    const unsigned short* kbase = kT + ((size_t)b * HW + ln) * C4;
    // V row pointers: A layout row m = c-within-tile
    const unsigned short* vbase[4];
#pragma unroll
    for (int tt = 0; tt < 4; ++tt)
        vbase[tt] = vmat + ((size_t)b * CC + wv * 64 + tt * 16 + ln) * HW;

    v8f acc[4];
#pragma unroll
    for (int tt = 0; tt < 4; ++tt)
#pragma unroll
        for (int e = 0; e < 8; ++e) acc[tt][e] = 0.f;

    float mrun = -3.0e38f;
    float lrun = 0.f;
    unsigned short* myp = &pT[wv][0];

    for (int j0 = 0; j0 < HW; j0 += 64) {
        // prefetch next block of K and V (global_prefetch_b8)
        if (j0 + 64 < HW) {
            __builtin_prefetch(kbase + (size_t)(j0 + 64) * C4, 0, 1);
            __builtin_prefetch(kbase + (size_t)(j0 + 96) * C4, 0, 1);
            __builtin_prefetch(vbase[0] + j0 + 64, 0, 1);
            __builtin_prefetch(vbase[2] + j0 + 64, 0, 1);
        }

        // ---- S^T: 4 j-tiles (rows j = j0+16t+m), cols i, contract c = 64 ----
        v8f s[4];
#pragma unroll
        for (int t = 0; t < 4; ++t) {
            const unsigned short* kr = kbase + (size_t)(j0 + 16 * t) * C4;
            FragB ka;
            v8f z = {0.f, 0.f, 0.f, 0.f, 0.f, 0.f, 0.f, 0.f};
            // c0 = 0   (A layout: u0 -> K = c0+8*hf.., u1 -> K = c0+16+8*hf..)
            ka.u[0] = *(const uint4*)(kr + 8 * hf);
            ka.u[1] = *(const uint4*)(kr + 16 + 8 * hf);
            z = wmma_bf16(ka.f, qb0.f, z);
            // c0 = 32
            ka.u[0] = *(const uint4*)(kr + 32 + 8 * hf);
            ka.u[1] = *(const uint4*)(kr + 48 + 8 * hf);
            s[t] = wmma_bf16(ka.f, qb1.f, z);
        }

        // ---- fp32 online softmax over j (rows of S^T) ----
        float mb = s[0][0];
#pragma unroll
        for (int t = 0; t < 4; ++t)
#pragma unroll
            for (int e = 0; e < 8; ++e) mb = fmaxf(mb, s[t][e]);
        mb = fmaxf(mb, __shfl_xor(mb, 16));          // combine halves (same i)
        const float mnew  = fmaxf(mrun, mb);
        const float alpha = exp2f((mrun - mnew) * L2E);

        float lb = 0.f;
#pragma unroll
        for (int t = 0; t < 4; ++t)
#pragma unroll
            for (int e = 0; e < 8; ++e) {
                s[t][e] = exp2f((s[t][e] - mnew) * L2E);   // P in place
                lb += s[t][e];
            }
        lb  += __shfl_xor(lb, 16);
        lrun = lrun * alpha + lb;
        mrun = mnew;

#pragma unroll
        for (int tt = 0; tt < 4; ++tt)
#pragma unroll
            for (int e = 0; e < 8; ++e) acc[tt][e] *= alpha;

        // ---- P (C/D layout) -> bf16 -> wave-private LDS pT[i][j] ----
        // lane holds i = ln; tile t, elem e -> j = 16t + 8*hf + e (contiguous)
#pragma unroll
        for (int t = 0; t < 4; ++t) {
            uint4 pw;
            pw.x = f2bf2(s[t][0], s[t][1]); pw.y = f2bf2(s[t][2], s[t][3]);
            pw.z = f2bf2(s[t][4], s[t][5]); pw.w = f2bf2(s[t][6], s[t][7]);
            *(uint4*)(myp + ln * 72 + 16 * t + 8 * hf) = pw;
        }

        // ---- read back as B fragments: lane n = i, u0 -> j = 32s+8*hf.., u1 -> j = 32s+16+8*hf..
        FragB pb0, pb1;
        pb0.u[0] = *(const uint4*)(myp + ln * 72 + 8 * hf);
        pb0.u[1] = *(const uint4*)(myp + ln * 72 + 16 + 8 * hf);
        pb1.u[0] = *(const uint4*)(myp + ln * 72 + 32 + 8 * hf);
        pb1.u[1] = *(const uint4*)(myp + ln * 72 + 48 + 8 * hf);

        // ---- O^T[c,i] += V(16c x 64j) x P^T(64j x 16i), 4 channel tiles ----
#pragma unroll
        for (int tt = 0; tt < 4; ++tt) {
            FragB va;
            const unsigned short* vr = vbase[tt] + j0;
            va.u[0] = *(const uint4*)(vr + 8 * hf);
            va.u[1] = *(const uint4*)(vr + 16 + 8 * hf);
            acc[tt] = wmma_bf16(va.f, pb0.f, acc[tt]);
            va.u[0] = *(const uint4*)(vr + 32 + 8 * hf);
            va.u[1] = *(const uint4*)(vr + 48 + 8 * hf);
            acc[tt] = wmma_bf16(va.f, pb1.f, acc[tt]);
        }
    }

    // ---- epilogue: normalize, gamma-scale, add residual, store (B,C,HW) ----
    const float gm = gamp[0] / lrun;   // lrun uniform across halves (same i)
#pragma unroll
    for (int tt = 0; tt < 4; ++tt) {
        const int cb = wv * 64 + tt * 16 + 8 * hf;   // C/D layout: m = e + 8*hf
#pragma unroll
        for (int e = 0; e < 8; ++e) {
            const size_t idx = ((size_t)b * CC + cb + e) * HW + i0 + ln;
            outp[idx] = fmaf(gm, acc[tt][e], x1r[idx]);
        }
    }
}

// ---------------------------------------------------------------------------
extern "C" void kernel_launch(void* const* d_in, const int* in_sizes, int n_in,
                              void* d_out, int out_size, void* d_ws, size_t ws_size,
                              hipStream_t stream)
{
    const float* x1  = (const float*)d_in[0];
    const float* x2  = (const float*)d_in[1];
    const float* x3  = (const float*)d_in[2];
    const float* w1  = (const float*)d_in[3];
    const float* b1  = (const float*)d_in[4];
    const float* g1  = (const float*)d_in[5];
    const float* be1 = (const float*)d_in[6];
    const float* m1  = (const float*)d_in[7];
    const float* v1  = (const float*)d_in[8];
    const float* w2  = (const float*)d_in[9];
    const float* b2  = (const float*)d_in[10];
    const float* g2  = (const float*)d_in[11];
    const float* be2 = (const float*)d_in[12];
    const float* m2  = (const float*)d_in[13];
    const float* v2  = (const float*)d_in[14];
    const float* w3  = (const float*)d_in[15];
    const float* b3  = (const float*)d_in[16];
    const float* g3  = (const float*)d_in[17];
    const float* be3 = (const float*)d_in[18];
    const float* m3  = (const float*)d_in[19];
    const float* v3  = (const float*)d_in[20];
    const float* gam = (const float*)d_in[21];

    // workspace: qT (2MB) | kT (2MB) | V (8MB), all bf16
    unsigned short* qT = (unsigned short*)d_ws;
    unsigned short* kT = qT + (size_t)BB * HW * C4;
    unsigned short* vm = kT + (size_t)BB * HW * C4;

    cbr_qk<<<dim3(HW / 4, BB), 256, 0, stream>>>(x1, w1, b1, g1, be1, m1, v1, qT);
    cbr_qk<<<dim3(HW / 4, BB), 256, 0, stream>>>(x2, w2, b2, g2, be2, m2, v2, kT);
    cbr_v <<<dim3(HW / 256, CC, BB), 256, 0, stream>>>(x3, w3, b3, g3, be3, m3, v3, vm);

    attn_fused<<<dim3(HW / 16, BB), 128, 0, stream>>>(qT, kT, vm, x1, gam, (float*)d_out);
}
